// acLSTM_11862699671613
// MI455X (gfx1250) — matrix operand order repaired and checked by reading, not compiled, not run on hardware
//
#include <hip/hip_runtime.h>
#include <stdint.h>

// ---------------------------------------------------------------------------
// acLSTM on gfx1250: persistent WMMA kernel.
//   - weights converted once to bf16 (L2-resident, ~43MB; L2 = 192MB)
//   - v_wmma_f32_16x16x32_bf16, M=16 gate rows, N=32 batch, K accumulated
//   - wave = K-eighth computing all 4 gates: B fragments loaded once per WG
//     (activation L2 traffic cut 4x), 8 independent WMMAs per k-step
//   - compile-time K trip counts -> fully unrolled, clause-batched b128 loads
//   - c-state held in registers across all 100 timesteps
//   - one grid barrier per step + producer-flag for the fed-back input
// ---------------------------------------------------------------------------

typedef __attribute__((ext_vector_type(8)))  float        v8f;
typedef __attribute__((ext_vector_type(16))) __bf16       v16bf;
typedef __attribute__((ext_vector_type(4)))  unsigned int u32x4;

namespace {
constexpr int kH    = 1024;
constexpr int kB    = 32;
constexpr int kT    = 100;
constexpr int kIn   = 132;
constexpr int kInP  = 160;     // input K padded to multiple of 32
constexpr int kOut  = 132;
constexpr int kKc0  = kInP + kH;   // 1184: [x | h] concatenated K for cell 0
constexpr int kKc12 = 2 * kH;      // 2048: [h_lower | h_self] for cells 1,2
constexpr int kNWG  = 192;         // 3 cells x 64 h-tiles
}

union ABFrag { v16bf bf; u32x4 q[2]; };

// A fragment (16x32 bf16, MxK): lane holds row M=lane&15; K-octets at
// hi*8 and 16+hi*8 (hi = lane>>4).  Two 16B loads from a row-major row.
__device__ __forceinline__ v16bf load_afrag(const unsigned short* p, int hi) {
  ABFrag a;
  a.q[0] = *reinterpret_cast<const u32x4*>(p + hi * 8);
  a.q[1] = *reinterpret_cast<const u32x4*>(p + 16 + hi * 8);
  return a.bf;
}
// B fragment (32x16 bf16, KxN): lane holds col N=lane&15; K-octets at
// hi*8 and 16+hi*8 (hi = lane>>4).  Two 16B loads from activation row n.
__device__ __forceinline__ v16bf load_bfrag(const unsigned short* p, int hi) {
  const unsigned short* q = p + hi * 8;
  ABFrag b;
  b.q[0] = *reinterpret_cast<const u32x4*>(q);
  b.q[1] = *reinterpret_cast<const u32x4*>(q + 16);
  return b.bf;
}

__device__ __forceinline__ v8f wmma_bf16(v16bf a, v16bf b, v8f c) {
  return __builtin_amdgcn_wmma_f32_16x16x32_bf16(false, a, false, b,
                                                 (short)0, c, false, false);
}

// NS k-steps for ALL FOUR gates x two N-tiles. B fragments loaded once and
// reused by 4 gates; 8 independent WMMAs per k-step. Fully unrolled.
template <int NS>
__device__ __forceinline__ void gemm4(const unsigned short* __restrict__ wr,
                                      size_t gstride,
                                      const unsigned short* __restrict__ b0,
                                      const unsigned short* __restrict__ b1,
                                      int hi, v8f (&acc)[8])
{
  #pragma unroll
  for (int s = 0; s < NS; ++s) {
    const int k0 = s * 32;
    v16bf bb0 = load_bfrag(b0 + k0, hi);
    v16bf bb1 = load_bfrag(b1 + k0, hi);
    #pragma unroll
    for (int gg = 0; gg < 4; ++gg) {
      v16bf a = load_afrag(wr + gstride * gg + k0, hi);
      acc[2 * gg + 0] = wmma_bf16(a, bb0, acc[2 * gg + 0]);
      acc[2 * gg + 1] = wmma_bf16(a, bb1, acc[2 * gg + 1]);
    }
  }
}

__device__ __forceinline__ unsigned short f2bf(float f) {  // RNE f32 -> bf16
  unsigned u = __float_as_uint(f);
  u += 0x7FFFu + ((u >> 16) & 1u);
  return (unsigned short)(u >> 16);
}
__device__ __forceinline__ float sigm(float x) { return 1.0f / (1.0f + __expf(-x)); }
__device__ __forceinline__ float tanhfast(float x) {
  return 1.0f - 2.0f / (__expf(2.0f * x) + 1.0f);  // stable at +/-inf
}

// Device-wide sense-reversing barrier: bar[0]=count, bar[1]=generation.
__device__ __forceinline__ void gbar(unsigned* bar, unsigned nwg) {
  __syncthreads();
  if (threadIdx.x == 0) {
    __threadfence();
    unsigned g = __hip_atomic_load(bar + 1, __ATOMIC_RELAXED, __HIP_MEMORY_SCOPE_AGENT);
    unsigned a = __hip_atomic_fetch_add(bar, 1u, __ATOMIC_ACQ_REL, __HIP_MEMORY_SCOPE_AGENT);
    if (a == nwg - 1u) {
      __hip_atomic_store(bar, 0u, __ATOMIC_RELAXED, __HIP_MEMORY_SCOPE_AGENT);
      __hip_atomic_store(bar + 1, g + 1u, __ATOMIC_RELEASE, __HIP_MEMORY_SCOPE_AGENT);
    } else {
      while (__hip_atomic_load(bar + 1, __ATOMIC_ACQUIRE, __HIP_MEMORY_SCOPE_AGENT) == g)
        __builtin_amdgcn_s_sleep(2);
    }
  }
  __syncthreads();
}

// ---------------------------------------------------------------------------
// One-time weight conversion: dst[m][0:Kxp] = pad(Wx row m), dst[m][Kxp:] = Wh.
// ---------------------------------------------------------------------------
__global__ void conv_w(const float* __restrict__ Wx, const float* __restrict__ Wh,
                       unsigned short* __restrict__ dst,
                       int rows, int srcRows, int Kx, int Kxp, int Kh)
{
  const long long Ktot = (long long)Kxp + Kh;
  const long long n = (long long)rows * Ktot;
  for (long long i = (long long)blockIdx.x * blockDim.x + threadIdx.x; i < n;
       i += (long long)gridDim.x * blockDim.x) {
    const long long m = i / Ktot;
    const long long k = i - m * Ktot;
    float v;
    if (m >= srcRows)  v = 0.f;
    else if (k < Kxp)  v = (k < Kx) ? Wx[m * Kx + k] : 0.f;
    else               v = Wh[m * Kh + (k - Kxp)];
    dst[i] = f2bf(v);
  }
}

// ---------------------------------------------------------------------------
// One-time state init: fused biases, padded decoder bias, x_0, zero h, flags.
// ---------------------------------------------------------------------------
__global__ void init_misc(const float* __restrict__ bi1, const float* __restrict__ bh1,
                          const float* __restrict__ bi2, const float* __restrict__ bh2,
                          const float* __restrict__ bi3, const float* __restrict__ bh3,
                          const float* __restrict__ bdin, const float* __restrict__ seq,
                          float* __restrict__ bsum, float* __restrict__ bdec,
                          unsigned short* __restrict__ xbuf,
                          unsigned short* __restrict__ hbuf,
                          unsigned* __restrict__ bar)
{
  const int i = blockIdx.x * blockDim.x + threadIdx.x;
  if (i < 12288) {                               // bsum[3][4096] = b_ih + b_hh
    const int cc = i >> 12, m = i & 4095;
    const float* bi = (cc == 0) ? bi1 : ((cc == 1) ? bi2 : bi3);
    const float* bh = (cc == 0) ? bh1 : ((cc == 1) ? bh2 : bh3);
    bsum[i] = bi[m] + bh[m];
  } else if (i < 12432) {                        // bdec[144] (padded)
    const int m = i - 12288;
    bdec[m] = (m < kOut) ? bdin[m] : 0.f;
  } else if (i < 17552) {                        // xbuf[32][160] = frame_0 (padded)
    const int k = i - 12432;
    const int b = k / kInP, kk = k - b * kInP;
    xbuf[k] = f2bf((kk < kIn) ? seq[(size_t)b * (kT * kIn) + kk] : 0.f);
  } else if (i < 214160) {                       // hbuf[2][3][32][1024] = 0
    hbuf[i - 17552] = 0;
  } else if (i < 214163) {                       // barrier count/gen + xcnt
    bar[i - 214160] = 0u;
  }
}

// ---------------------------------------------------------------------------
// Persistent time-loop kernel. 192 WGs x 256 threads (8 wave32s).
//   block = (cell c = bid>>6, h-tile j = bid&63); wave = 256-wide K chunk,
//   computing all 4 gates for it (B fragments loaded once per WG).
// ---------------------------------------------------------------------------
__global__ __launch_bounds__(256, 1) void lstm_persist(
    const unsigned short* __restrict__ W0, const unsigned short* __restrict__ W1,
    const unsigned short* __restrict__ W2, const unsigned short* __restrict__ Wd,
    const float* __restrict__ bsum, const float* __restrict__ bdec,
    unsigned short* __restrict__ xbuf, unsigned short* __restrict__ hbuf,
    const float* __restrict__ seq, float* __restrict__ dout,
    unsigned* __restrict__ bar)
{
  __shared__ float lds[8][4][2][8][32];          // [wave][gate][Ntile][r][lane]
  const int tid  = threadIdx.x;
  const int lane = tid & 31;
  const int w    = tid >> 5;                     // 0..7: K-chunk index
  const int hi   = lane >> 4;
  const int ml   = lane & 15;
  const int bid  = blockIdx.x;
  const int c    = bid >> 6;                     // 0..2
  const int j    = bid & 63;                     // h-tile

  const unsigned short* Wc = (c == 0) ? W0 : ((c == 1) ? W1 : W2);
  const int Kc   = (c == 0) ? kKc0 : kKc12;
  const unsigned short* wbase = Wc + (size_t)(j * 16 + ml) * (size_t)Kc;  // gate 0 row
  const size_t gstride = (size_t)kH * (size_t)Kc;                         // gate->gate
  const unsigned short* wrowD = Wd + (size_t)(bid * 16 + ml) * (size_t)kH + (size_t)w * 128;

  unsigned* xcnt = bar + 2;                      // fed-back-input ready flag
  float creg[2] = {0.f, 0.f};                    // cell state, registers, all T

  for (int t = 0; t < kT; ++t) {
    const int p = t & 1;                         // read parity for h buffers
    // ---------------- phase 1: the three LSTM cells -----------------------
    {
      const unsigned short* act0 =
          (c == 0) ? xbuf : (hbuf + ((size_t)p * 3 + (size_t)(c - 1)) * (size_t)(kB * kH));
      const int pitch0 = (c == 0) ? kInP : kH;
      const unsigned short* act1 = hbuf + ((size_t)p * 3 + (size_t)c) * (size_t)(kB * kH);
      const unsigned short* bp0 = act1 + (size_t)ml * kH;          // h_l rows
      const unsigned short* bp1 = act1 + (size_t)(16 + ml) * kH;
      const unsigned short* ap0 = act0 + (size_t)ml * pitch0;      // x / h_{l-1} rows
      const unsigned short* ap1 = act0 + (size_t)(16 + ml) * pitch0;

      v8f acc[8];
      #pragma unroll
      for (int i = 0; i < 8; ++i) acc[i] = (v8f){0, 0, 0, 0, 0, 0, 0, 0};

      if (c == 0) {
        if (w < 4) {                 // recurrent part: K chunk w*256 of h_0
          gemm4<8>(wbase + kInP + w * 256, gstride,
                   bp0 + w * 256, bp1 + w * 256, hi, acc);
        } else if (w == 4) {         // x part: waits for the fed-back input
          const unsigned want = 9u * (unsigned)t;
          while (__hip_atomic_load(xcnt, __ATOMIC_ACQUIRE, __HIP_MEMORY_SCOPE_AGENT) < want)
            __builtin_amdgcn_s_sleep(1);
          gemm4<5>(wbase, gstride, ap0, ap1, hi, acc);
        }                            // waves 5..7: zero partials
      } else {
        if (w < 4)                   // h_{l-1} part
          gemm4<8>(wbase + w * 256, gstride,
                   ap0 + w * 256, ap1 + w * 256, hi, acc);
        else                         // h_l part
          gemm4<8>(wbase + kH + (w - 4) * 256, gstride,
                   bp0 + (w - 4) * 256, bp1 + (w - 4) * 256, hi, acc);
      }

      #pragma unroll
      for (int gg = 0; gg < 4; ++gg) {
        #pragma unroll
        for (int r = 0; r < 8; ++r) {
          lds[w][gg][0][r][lane] = acc[2 * gg + 0][r];
          lds[w][gg][1][r][lane] = acc[2 * gg + 1][r];
        }
      }
      __syncthreads();
      // fused LSTM elementwise; c kept in registers, h written as bf16
      unsigned short* hout = hbuf + ((size_t)(1 - p) * 3 + (size_t)c) * (size_t)(kB * kH);
      #pragma unroll
      for (int q = 0; q < 2; ++q) {
        const int slot = tid + q * 256;
        const int nt = slot >> 8, r = (slot >> 5) & 7, ln = slot & 31;
        const int n = nt * 16 + (ln & 15);
        const int m = j * 16 + r + ((ln >> 4) << 3);
        const float* bb = bsum + (size_t)c * 4 * kH + m;
        float gsum[4];
        #pragma unroll
        for (int G = 0; G < 4; ++G) {
          float s = bb[G * kH];
          #pragma unroll
          for (int ww = 0; ww < 8; ++ww) s += lds[ww][G][nt][r][ln];
          gsum[G] = s;
        }
        const float cn = sigm(gsum[1]) * creg[q] + sigm(gsum[0]) * tanhfast(gsum[2]);
        creg[q] = cn;
        hout[(size_t)n * kH + m] = f2bf(sigm(gsum[3]) * tanhfast(cn));
      }
    }
    gbar(bar, kNWG);                             // nh2 visible device-wide
    // ---------------- phase 2: decoder + next-step input ------------------
    if (bid < 9) {
      const unsigned short* act = hbuf + ((size_t)(1 - p) * 3 + 2) * (size_t)(kB * kH);
      v8f d0 = {0, 0, 0, 0, 0, 0, 0, 0};
      v8f d1 = {0, 0, 0, 0, 0, 0, 0, 0};
      const int kb = w * 128;                    // each of 8 waves: 128 K cols
      const unsigned short* bp0 = act + (size_t)ml * kH + kb;
      const unsigned short* bp1 = act + (size_t)(16 + ml) * kH + kb;
      #pragma unroll
      for (int s = 0; s < 4; ++s) {
        const int k0 = s * 32;
        v16bf a   = load_afrag(wrowD + k0, hi);
        v16bf bb0 = load_bfrag(bp0 + k0, hi);
        v16bf bb1 = load_bfrag(bp1 + k0, hi);
        d0 = wmma_bf16(a, bb0, d0);
        d1 = wmma_bf16(a, bb1, d1);
      }
      #pragma unroll
      for (int r = 0; r < 8; ++r) {
        lds[w][0][0][r][lane] = d0[r];
        lds[w][0][1][r][lane] = d1[r];
      }
      __syncthreads();
      #pragma unroll
      for (int q = 0; q < 2; ++q) {
        const int slot = tid + q * 256;
        const int nt = slot >> 8, r = (slot >> 5) & 7, ln = slot & 31;
        const int m = bid * 16 + r + ((ln >> 4) << 3);
        const int n = nt * 16 + (ln & 15);
        if (m < kOut) {
          float v = bdec[m];
          #pragma unroll
          for (int ww = 0; ww < 8; ++ww) v += lds[ww][0][nt][r][ln];
          dout[(size_t)n * (kT * kOut) + (size_t)t * kOut + m] = v;
          if (t < kT - 1) {                      // scheduled sampling
            const bool gt = (((t + 1) % 10) < 5);
            const float xv = gt ? seq[(size_t)n * (kT * kIn) + (size_t)(t + 1) * kIn + m] : v;
            xbuf[(size_t)n * kInP + m] = f2bf(xv);
          }
        }
      }
      __syncthreads();                           // xbuf writes done block-wide
      if (tid == 0) {
        __threadfence();
        __hip_atomic_fetch_add(xcnt, 1u, __ATOMIC_RELEASE, __HIP_MEMORY_SCOPE_AGENT);
      }
    }
  }
}

// ---------------------------------------------------------------------------
extern "C" void kernel_launch(void* const* d_in, const int* in_sizes, int n_in,
                              void* d_out, int out_size, void* d_ws, size_t ws_size,
                              hipStream_t stream)
{
  (void)in_sizes; (void)n_in; (void)out_size; (void)ws_size;
  const float* seq  = (const float*)d_in[0];
  const float* Wih1 = (const float*)d_in[1];
  const float* Whh1 = (const float*)d_in[2];
  const float* bih1 = (const float*)d_in[3];
  const float* bhh1 = (const float*)d_in[4];
  const float* Wih2 = (const float*)d_in[5];
  const float* Whh2 = (const float*)d_in[6];
  const float* bih2 = (const float*)d_in[7];
  const float* bhh2 = (const float*)d_in[8];
  const float* Wih3 = (const float*)d_in[9];
  const float* Whh3 = (const float*)d_in[10];
  const float* bih3 = (const float*)d_in[11];
  const float* bhh3 = (const float*)d_in[12];
  const float* Wdec = (const float*)d_in[13];
  const float* bdin = (const float*)d_in[14];
  float* dout = (float*)d_out;
  uint8_t* ws = (uint8_t*)d_ws;

  size_t off = 0;
  auto carve = [&](size_t bytes) { size_t r = off; off += (bytes + 255) & ~(size_t)255; return r; };
  unsigned short* W0   = (unsigned short*)(ws + carve((size_t)4096 * 1184 * 2));
  unsigned short* W1   = (unsigned short*)(ws + carve((size_t)4096 * 2048 * 2));
  unsigned short* W2   = (unsigned short*)(ws + carve((size_t)4096 * 2048 * 2));
  unsigned short* Wd   = (unsigned short*)(ws + carve((size_t)144 * 1024 * 2));
  float*          bsum = (float*)(ws + carve((size_t)12288 * 4));
  float*          bdec = (float*)(ws + carve((size_t)144 * 4));
  unsigned short* xbuf = (unsigned short*)(ws + carve((size_t)32 * 160 * 2));
  unsigned short* hbuf = (unsigned short*)(ws + carve((size_t)2 * 3 * 32 * 1024 * 2));
  unsigned*       bar  = (unsigned*)(ws + carve(256));

  // one-time bf16 weight staging (L2-resident thereafter)
  conv_w<<<2048, 256, 0, stream>>>(Wih1, Whh1, W0, 4096, 4096, 132, 160, 1024);
  conv_w<<<2048, 256, 0, stream>>>(Wih2, Whh2, W1, 4096, 4096, 1024, 1024, 1024);
  conv_w<<<2048, 256, 0, stream>>>(Wih3, Whh3, W2, 4096, 4096, 1024, 1024, 1024);
  conv_w<<<256, 256, 0, stream>>>(Wdec, Wdec, Wd, 144, 132, 1024, 1024, 0);
  init_misc<<<(214163 + 255) / 256, 256, 0, stream>>>(bih1, bhh1, bih2, bhh2, bih3, bhh3,
                                                      bdin, seq, bsum, bdec, xbuf, hbuf, bar);
  // persistent time loop
  lstm_persist<<<kNWG, 256, 0, stream>>>(W0, W1, W2, Wd, bsum, bdec, xbuf, hbuf,
                                         seq, dout, bar);
}
